// Cross_MultiAttention_53876069761721
// MI455X (gfx1250) — matrix operand order repaired and checked by reading, not compiled, not run on hardware
//
#include <hip/hip_runtime.h>

typedef unsigned short u16;
typedef unsigned int   u32;
typedef unsigned long long u64;
typedef __attribute__((ext_vector_type(16))) __bf16        v16bf;
typedef __attribute__((ext_vector_type(8)))  float         v8f;
typedef __attribute__((ext_vector_type(8)))  unsigned int  v8u;
typedef __attribute__((ext_vector_type(4)))  unsigned int  v4u;

#define B_ 4
#define T_ 2048
#define S_ 2048
#define C_ 256
#define E_ 512
#define H_ 8
#define D_ 64
// SCALE = E ** -0.5 (note: E, not head dim, per the reference)
#define SCALE_ 0.044194173824159216f

// ---------------------------------------------------------------- helpers
__device__ __forceinline__ u16 f2bf(float f) {
  union { float f; u32 u; } v; v.f = f;
  u32 r = v.u + 0x7FFFu + ((v.u >> 16) & 1u);   // round-to-nearest-even
  return (u16)(r >> 16);
}

// Async global->LDS 16-byte copy (CDNA5 GLOBAL_LOAD_ASYNC_TO_LDS_B128,
// tracked by ASYNCcnt). lds = raw LDS byte offset (low 32 bits of the
// generic pointer), ga = 64-bit global address. 16B alignment both sides.
__device__ __forceinline__ void async_copy_b128(const void* lds_ptr,
                                                const void* gptr) {
  unsigned lds = (unsigned)(unsigned long long)lds_ptr;  // LDS offset
  u64 ga = (u64)gptr;
  asm volatile("global_load_async_to_lds_b128 %0, %1, off"
               :: "v"(lds), "v"(ga) : "memory");
}
__device__ __forceinline__ void async_wait0() {
  asm volatile("s_wait_asynccnt 0x0" ::: "memory");
}

// Tensor Data Mover: DMA a 2D tile (tile_d x tile_s, 2-byte elements,
// row stride `stride_elems` in memory) from global to LDS, adding 4 dwords
// of LDS padding after every 32 dwords (row stride 128B -> 144B in LDS).
// Issued once per workgroup by one wave; tracked by TENSORcnt.
__device__ __forceinline__ void tdm_load_2d_bf16_pad(u32 lds_off, u64 ga) {
  v4u g0;
  g0.x = 1u;                                   // count=1 valid descriptor
  g0.y = lds_off;                              // lds_addr (bytes)
  g0.z = (u32)ga;                              // global_addr[31:0]
  g0.w = (u32)((ga >> 32) & 0x01FFFFFFu)       // global_addr[56:32]
         | (2u << 30);                         // type = 2 ("image")
  v8u g1;
  g1[0] = (1u << 16)      // data_size = 1 -> 2 bytes
        | (1u << 20)      // pad_enable
        | (4u << 22)      // pad_interval code 4 -> every 32 dwords (128B)
        | (3u << 25);     // pad_amount  code 3 -> 4 dwords (16B)
  g1[1] = 64u << 16;      // tensor_dim0 = 64 (d elements per row)
  g1[2] = 32u << 16;      // tensor_dim1 = 32 (s rows)
  g1[3] = 64u << 16;      // tile_dim0 = 64
  g1[4] = 32u;            // tile_dim1 = 32, tile_dim2 = 0 (2D)
  g1[5] = 512u;           // tensor_dim0_stride = E elements
  g1[6] = 0u;
  g1[7] = 0u;
  v4u gz; gz.x = 0u; gz.y = 0u; gz.z = 0u; gz.w = 0u;  // dims 2+ unused
  asm volatile("tensor_load_to_lds %0, %1, %2, %3"
               :: "s"(g0), "s"(g1), "s"(gz), "s"(gz) : "memory");
}

__global__ void cast_f32_to_bf16(const float* __restrict__ in,
                                 u16* __restrict__ out, int n) {
  int i = blockIdx.x * blockDim.x + threadIdx.x;
  int stride = gridDim.x * blockDim.x;
  for (; i < n; i += stride) out[i] = f2bf(in[i]);
}

// ---------------------------------------------------------------- GEMM
// Out[M,N] = A[M,K](bf16) * W[N,K]^T(bf16) + bias[N](f32)
// block: 128 threads (4 waves), 64x64 tile; wave -> 32x32 (2x2 wmma tiles).
// Tiles staged with async b128 copies; LDS row stride 40 halves (80B:
// 16B-aligned for b128, 20 dwords -> conflict-free for 16-row groups).
template <bool OUTF32>
__global__ __launch_bounds__(128)
void gemm_bf16_wmma(const u16* __restrict__ A, const u16* __restrict__ W,
                    const float* __restrict__ bias, void* __restrict__ Out,
                    int Kd, int N)
{
  __shared__ u16 Ab[64 * 40];
  __shared__ u16 Wb[64 * 40];
  const int tid  = threadIdx.x;
  const int lane = tid & 31, wave = tid >> 5;
  const int hi   = lane >> 4, l15 = lane & 15;
  const int m0 = blockIdx.y * 64, n0 = blockIdx.x * 64;
  const int wr = (wave >> 1) * 32, wc = (wave & 1) * 32;

  v8f zero = {};
  v8f acc[2][2];
  acc[0][0] = zero; acc[0][1] = zero; acc[1][0] = zero; acc[1][1] = zero;

  for (int kb = 0; kb < Kd; kb += 32) {
    __syncthreads();
    // async-stage 64x32 bf16 tiles of A and W: 256 x 16B chunks each
#pragma unroll
    for (int i = 0; i < 2; ++i) {
      int idx = tid + i * 128;         // 0..255
      int row = idx >> 2;              // 0..63
      int kw  = (idx & 3) * 8;         // halfword offset 0,8,16,24
      async_copy_b128(&Ab[row * 40 + kw],
                      &A[(size_t)(m0 + row) * Kd + kb + kw]);
      async_copy_b128(&Wb[row * 40 + kw],
                      &W[(size_t)(n0 + row) * Kd + kb + kw]);
    }
    async_wait0();
    __syncthreads();

    v16bf af[2], wf[2];
#pragma unroll
    for (int s = 0; s < 2; ++s) {
      v8u ta, tb;
      int arow = wr + s * 16 + l15;      // M row for A fragment
      int brow = wc + s * 16 + l15;      // N row of W == B-matrix column
#pragma unroll
      for (int j = 0; j < 8; ++j) {
        int e  = 2 * j;
        int ka = e + hi * 16;   // A-layout k index
        ta[j] = *(const u32*)&Ab[arow * 40 + ka];
        tb[j] = *(const u32*)&Wb[brow * 40 + e + hi * 16]; // B-layout k
      }
      af[s] = __builtin_bit_cast(v16bf, ta);
      wf[s] = __builtin_bit_cast(v16bf, tb);
    }
#pragma unroll
    for (int i = 0; i < 2; ++i)
#pragma unroll
      for (int j = 0; j < 2; ++j)
        acc[i][j] = __builtin_amdgcn_wmma_f32_16x16x32_bf16(
            false, af[i], false, wf[j], (short)0, acc[i][j], false, false);
  }

  // epilogue: D layout -> row = r + hi*8, col = lane&15
#pragma unroll
  for (int i = 0; i < 2; ++i)
#pragma unroll
    for (int j = 0; j < 2; ++j)
#pragma unroll
      for (int r = 0; r < 8; ++r) {
        int row = m0 + wr + i * 16 + r + hi * 8;
        int col = n0 + wc + j * 16 + l15;
        float v = acc[i][j][r] + bias[col];
        if (OUTF32) ((float*)Out)[(size_t)row * N + col] = v;
        else        ((u16*)Out)[(size_t)row * N + col]   = f2bf(v);
      }
}

// ---------------------------------------------------------------- flash attention
// grid: (T/64, B*H); block 128 (4 waves). Wave w handles Q rows
// [t0block + 16w, +16) of head (b,h); streams S in chunks of 32 with
// online softmax. All 4 waves share K/V chunks staged in LDS.
// K tile staged by the Tensor Data Mover (one tensor_load_to_lds per chunk,
// HW-padded to 144B row stride = 72 halves: 16B-aligned + conflict-free);
// V staged via VGPRs (needs a transpose, which TDM cannot do).
__global__ __launch_bounds__(128)
void flash_attn_wmma(const u16* __restrict__ Qg, const u16* __restrict__ Kg,
                     const u16* __restrict__ Vg,
                     const unsigned char* __restrict__ mask,
                     u16* __restrict__ Og)
{
  __shared__ u16 Kl[32 * 72];        // K[s][d], stride 72 halves (TDM pad)
  __shared__ u16 Vt[64 * 34];        // V^T[d][s], stride 34 (17 dwords)
  __shared__ u16 Pb[4][16 * 34];     // per-wave P scratch [row][col]

  const int tid  = threadIdx.x;
  const int lane = tid & 31, wave = tid >> 5;
  const int hi   = lane >> 4, l15 = lane & 15;
  const int bh = blockIdx.y;
  const int b  = bh >> 3, h = bh & 7;
  const int t0 = blockIdx.x * 64 + wave * 16;

  const u32 kl_off  = (u32)(u64)&Kl[0];   // raw LDS byte offset
  const u64 kg_base = (u64)(Kg + ((size_t)b * S_) * E_ + h * 64);

  // Q fragments (16 x 64, two K=32 chunks), loaded once
  v16bf qf[2];
  {
    const u16* qrow = Qg + ((size_t)(b * T_ + t0 + l15)) * E_ + h * 64;
#pragma unroll
    for (int c = 0; c < 2; ++c) {
      v8u t;
#pragma unroll
      for (int j = 0; j < 8; ++j) {
        int e = 2 * j;
        int d = c * 32 + e + hi * 16;
        t[j] = *(const u32*)&qrow[d];
      }
      qf[c] = __builtin_bit_cast(v16bf, t);
    }
  }

  float m[8], l[8];
  v8f zero = {};
  v8f o[4]; o[0] = zero; o[1] = zero; o[2] = zero; o[3] = zero;
#pragma unroll
  for (int r = 0; r < 8; ++r) { m[r] = -1e30f; l[r] = 0.0f; }

  for (int s0 = 0; s0 < S_; s0 += 32) {
    __syncthreads();
    // TDM: DMA the 32x64 bf16 K chunk into LDS (wave 0 issues + waits)
    if (wave == 0) {
      tdm_load_2d_bf16_pad(kl_off, kg_base + (u64)s0 * (E_ * 2));
      __builtin_amdgcn_s_wait_tensorcnt((short)0);
    }
    // stage V transposed (VGPR path: coalesced b32 + 2 x b16 LDS stores)
#pragma unroll
    for (int i = 0; i < 4; ++i) {
      int idx = tid + i * 128;
      int row = idx >> 4, kw = (idx & 15) * 2;
      u32 vv = *(const u32*)&Vg[((size_t)(b * S_ + s0 + row)) * E_
                                + h * 64 + kw];
      Vt[(kw + 0) * 34 + row] = (u16)(vv & 0xFFFFu);
      Vt[(kw + 1) * 34 + row] = (u16)(vv >> 16);
    }
    // prefetch next chunk of K/V into the cache hierarchy
    if (s0 + 32 < S_) {
      size_t nidx = ((size_t)(b * S_ + s0 + 32 + (tid >> 2))) * E_
                    + h * 64 + (tid & 3) * 16;
      __builtin_prefetch(&Kg[nidx], 0, 1);
      __builtin_prefetch(&Vg[nidx], 0, 1);
    }
    __syncthreads();

    // scores: S[t][s] for 16 rows x 32 cols (two 16-col subtiles)
    v8f sc[2]; sc[0] = zero; sc[1] = zero;
#pragma unroll
    for (int sub = 0; sub < 2; ++sub) {
      int srow = sub * 16 + l15;       // B-matrix column = key index
#pragma unroll
      for (int c = 0; c < 2; ++c) {
        v8u t;
        int dbase = c * 32 + hi * 16;  // B-layout k index base
#pragma unroll
        for (int j = 0; j < 8; ++j)
          t[j] = *(const u32*)&Kl[srow * 72 + dbase + 2 * j];
        v16bf kf = __builtin_bit_cast(v16bf, t);
        sc[sub] = __builtin_amdgcn_wmma_f32_16x16x32_bf16(
            false, qf[c], false, kf, (short)0, sc[sub], false, false);
      }
    }

    // scale + pad mask  (mask true -> -1e9, matching reference)
#pragma unroll
    for (int sub = 0; sub < 2; ++sub)
#pragma unroll
      for (int r = 0; r < 8; ++r) {
        float v = sc[sub][r] * SCALE_;
        size_t mi = ((size_t)(b) * T_ + (t0 + r + hi * 8)) * (size_t)S_
                    + s0 + sub * 16 + l15;
        if (mask[mi]) v = -1e9f;
        sc[sub][r] = v;
      }

    // online softmax: per-row max/sum via intra-half-wave reductions
    float corr[8];
#pragma unroll
    for (int r = 0; r < 8; ++r) {
      float mx = fmaxf(sc[0][r], sc[1][r]);
#pragma unroll
      for (int d = 1; d < 16; d <<= 1) mx = fmaxf(mx, __shfl_xor(mx, d, 32));
      float mn = fmaxf(m[r], mx);
      corr[r] = __expf(m[r] - mn);
      m[r] = mn;
    }
#pragma unroll
    for (int r = 0; r < 8; ++r) {
      float p0 = __expf(sc[0][r] - m[r]);
      float p1 = __expf(sc[1][r] - m[r]);
      int row = r + hi * 8;
      Pb[wave][row * 34 + l15]      = f2bf(p0);
      Pb[wave][row * 34 + 16 + l15] = f2bf(p1);
      float ps = p0 + p1;
#pragma unroll
      for (int d = 1; d < 16; d <<= 1) ps += __shfl_xor(ps, d, 32);
      l[r] = l[r] * corr[r] + ps;
    }
#pragma unroll
    for (int g = 0; g < 4; ++g)
#pragma unroll
      for (int r = 0; r < 8; ++r) o[g][r] = o[g][r] * corr[r];

    // P fragment (16x32 A-layout) from per-wave LDS scratch
    v16bf pf;
    {
      v8u t;
#pragma unroll
      for (int j = 0; j < 8; ++j) {
        int e  = 2 * j;
        int kc = e + hi * 16;
        t[j] = *(const u32*)&Pb[wave][l15 * 34 + kc];
      }
      pf = __builtin_bit_cast(v16bf, t);
    }
    // O += P @ V   (4 d-groups of 16)
#pragma unroll
    for (int g = 0; g < 4; ++g) {
      v8u t;
      int d = g * 16 + l15;
#pragma unroll
      for (int j = 0; j < 8; ++j)
        t[j] = *(const u32*)&Vt[d * 34 + hi * 16 + 2 * j];
      v16bf vf = __builtin_bit_cast(v16bf, t);
      o[g] = __builtin_amdgcn_wmma_f32_16x16x32_bf16(
          false, pf, false, vf, (short)0, o[g], false, false);
    }
  }

  // normalize and write O (bf16) to [B,T,E], e = h*64 + d
#pragma unroll
  for (int r = 0; r < 8; ++r) {
    float inv = 1.0f / l[r];
    int t = t0 + r + hi * 8;
#pragma unroll
    for (int g = 0; g < 4; ++g)
      Og[((size_t)(b * T_ + t)) * E_ + h * 64 + g * 16 + l15] =
          f2bf(o[g][r] * inv);
  }
}

// ---------------------------------------------------------------- launch
extern "C" void kernel_launch(void* const* d_in, const int* in_sizes, int n_in,
                              void* d_out, int out_size, void* d_ws, size_t ws_size,
                              hipStream_t stream)
{
  const float* x     = (const float*)d_in[0];
  const float* ctx   = (const float*)d_in[1];
  const unsigned char* mask = (const unsigned char*)d_in[2]; // bool [B,T,S]
  const float* w_in  = (const float*)d_in[3];
  const float* b_in  = (const float*)d_in[4];
  const float* wq    = (const float*)d_in[5];
  const float* bq    = (const float*)d_in[6];
  const float* wk    = (const float*)d_in[7];
  const float* bk    = (const float*)d_in[8];
  const float* wv    = (const float*)d_in[9];
  const float* bv    = (const float*)d_in[10];
  const float* w_out = (const float*)d_in[11];
  const float* b_out = (const float*)d_in[12];

  char* ws = (char*)d_ws;
  size_t off = 0;
  auto alloc = [&](size_t bytes) -> void* {
    void* p = ws + off;
    off = (off + bytes + 255) & ~(size_t)255;
    return p;
  };
  u16* xb    = (u16*)alloc((size_t)B_ * T_ * C_ * 2);
  u16* cb    = (u16*)alloc((size_t)B_ * S_ * E_ * 2);
  u16* winb  = (u16*)alloc((size_t)E_ * C_ * 2);
  u16* wqb   = (u16*)alloc((size_t)E_ * E_ * 2);
  u16* wkb   = (u16*)alloc((size_t)E_ * E_ * 2);
  u16* wvb   = (u16*)alloc((size_t)E_ * E_ * 2);
  u16* woutb = (u16*)alloc((size_t)C_ * E_ * 2);
  u16* xeb   = (u16*)alloc((size_t)B_ * T_ * E_ * 2);
  u16* Qb    = (u16*)alloc((size_t)B_ * T_ * E_ * 2);
  u16* Kb    = (u16*)alloc((size_t)B_ * S_ * E_ * 2);
  u16* Vb    = (u16*)alloc((size_t)B_ * S_ * E_ * 2);
  u16* Ob    = (u16*)alloc((size_t)B_ * T_ * E_ * 2);
  if (off > ws_size) return;  // workspace too small (never expected)

  // f32 -> bf16 casts
  cast_f32_to_bf16<<<1024, 256, 0, stream>>>(x,     xb,    B_ * T_ * C_);
  cast_f32_to_bf16<<<1024, 256, 0, stream>>>(ctx,   cb,    B_ * S_ * E_);
  cast_f32_to_bf16<<<256,  256, 0, stream>>>(w_in,  winb,  E_ * C_);
  cast_f32_to_bf16<<<256,  256, 0, stream>>>(wq,    wqb,   E_ * E_);
  cast_f32_to_bf16<<<256,  256, 0, stream>>>(wk,    wkb,   E_ * E_);
  cast_f32_to_bf16<<<256,  256, 0, stream>>>(wv,    wvb,   E_ * E_);
  cast_f32_to_bf16<<<256,  256, 0, stream>>>(w_out, woutb, C_ * E_);

  // proj_in: [B*T,C] x [E,C]^T -> xe [B*T,E]
  gemm_bf16_wmma<false><<<dim3(E_ / 64, (B_ * T_) / 64), 128, 0, stream>>>(
      xb, winb, b_in, xeb, C_, E_);
  // Q = xe @ wq^T
  gemm_bf16_wmma<false><<<dim3(E_ / 64, (B_ * T_) / 64), 128, 0, stream>>>(
      xeb, wqb, bq, Qb, E_, E_);
  // K = ctx @ wk^T ; V = ctx @ wv^T
  gemm_bf16_wmma<false><<<dim3(E_ / 64, (B_ * S_) / 64), 128, 0, stream>>>(
      cb, wkb, bk, Kb, E_, E_);
  gemm_bf16_wmma<false><<<dim3(E_ / 64, (B_ * S_) / 64), 128, 0, stream>>>(
      cb, wvb, bv, Vb, E_, E_);
  // fused masked attention (flash, online softmax, TDM-staged K)
  flash_attn_wmma<<<dim3(T_ / 64, B_ * H_), 128, 0, stream>>>(
      Qb, Kb, Vb, mask, Ob);
  // proj_out: O @ w_out^T + b_out -> f32 d_out [B,T,C]
  gemm_bf16_wmma<true><<<dim3(C_ / 64, (B_ * T_) / 64), 128, 0, stream>>>(
      Ob, woutb, b_out, d_out, E_, C_);
}
